// ModifiedTransformerBlock_32650341384412
// MI455X (gfx1250) — compile-verified, matrix operand order checked by reading, not run on hardware
//
#include <hip/hip_runtime.h>

typedef _Float16 h16;
typedef __attribute__((ext_vector_type(16))) _Float16 v16h;
typedef __attribute__((ext_vector_type(8)))  _Float16 v8h;
typedef __attribute__((ext_vector_type(8)))  float    v8f;
typedef __attribute__((ext_vector_type(2)))  float    v2f;
typedef __attribute__((ext_vector_type(4)))  int      i32x4;
typedef __attribute__((ext_vector_type(8)))  int      i32x8;

#define BB   256
#define TT   256
#define CC   384
#define HH   6
#define HSZ  64
#define PP   307
#define PPAD 320
#define BTN  (BB*TT)        // 65536 tokens
#define NQKV (HH*192)       // 1152

static __device__ inline v8f zero8(){ v8f z={0.f,0.f,0.f,0.f,0.f,0.f,0.f,0.f}; return z; }
static __device__ inline v16h cat8(v8h a, v8h b){
  return __builtin_shufflevector(a,b,0,1,2,3,4,5,6,7,8,9,10,11,12,13,14,15);
}

// ---------------------------------------------------------------------------
// TDM: DMA a [128 rows x 32 f16] tile (row stride K elems) from global memory
// into LDS with hardware padding: 16 DWORDs data + 4 DWORDs pad = 80B pitch.
// 2D tensor -> 2-group descriptor form (VADDR2/3 omitted = NULL).
// ---------------------------------------------------------------------------
static __device__ inline void tdm_load_tileA(const h16* gsrc, unsigned ldsOff, int K, int M){
  unsigned long long ga = (unsigned long long)(uintptr_t)gsrc;
  i32x4 g0;
  g0[0] = 1;                                        // count=1, user descriptor
  g0[1] = (int)ldsOff;                              // lds byte address
  g0[2] = (int)(ga & 0xffffffffull);                // global_addr[31:0]
  g0[3] = (int)((ga>>32) & 0x1ffffffull) | (2<<30); // global_addr[56:32] | type=2
  i32x8 g1;
  g1[0] = (1<<16) | (1<<20) | (3<<22) | (3<<25);    // data_size=2B, pad_en, ivl=16dw, amt=4dw
  g1[1] = (K & 0xffff) << 16;                       // tensor_dim0[15:0] (atomic_barrier=0)
  g1[2] = ((K>>16)&0xffff) | ((M & 0xffff)<<16);    // tensor_dim0[31:16] | tensor_dim1[15:0]
  g1[3] = ((M>>16)&0xffff) | (32<<16);              // tensor_dim1[31:16] | tile_dim0=32
  g1[4] = 128;                                      // tile_dim1=128, tile_dim2=0
  g1[5] = K;                                        // tensor_dim0_stride[31:0]
  g1[6] = 0;                                        // stride hi | dim1_stride lo
  g1[7] = 0;
  asm volatile("tensor_load_to_lds %0, %1" :: "s"(g0), "s"(g1) : "memory");
}

// ---------------------------------------------------------------------------
// RMSNorm: one block per row (C=384), 128 threads. Writes f32 and/or f16.
// ---------------------------------------------------------------------------
__global__ __launch_bounds__(128)
void rmsnorm_kernel(const float* __restrict__ x, const float* __restrict__ scale,
                    float* __restrict__ outF, h16* __restrict__ outH){
  const int row = blockIdx.x;
  const float* xr = x + (size_t)row*CC;
  const int t = threadIdx.x;
  float v0=xr[t], v1=xr[t+128], v2=xr[t+256];
  float ss = v0*v0 + v1*v1 + v2*v2;
  #pragma unroll
  for(int d=1;d<32;d<<=1) ss += __shfl_xor(ss,d,32);
  __shared__ float sb[4];
  if((t&31)==0) sb[t>>5]=ss;
  __syncthreads();
  float tot = sb[0]+sb[1]+sb[2]+sb[3];
  float inv = rsqrtf(tot*(1.0f/CC) + 1e-6f);
  float o0=v0*inv*scale[t], o1=v1*inv*scale[t+128], o2=v2*inv*scale[t+256];
  if(outF){ float* of=outF+(size_t)row*CC; of[t]=o0; of[t+128]=o1; of[t+256]=o2; }
  if(outH){ h16* oh=outH+(size_t)row*CC; oh[t]=(h16)o0; oh[t+128]=(h16)o1; oh[t+256]=(h16)o2; }
}

// ---------------------------------------------------------------------------
// Weight packing / padding / swizzling into WMMA B-fragment order.
// ---------------------------------------------------------------------------
__global__ void pack_qkv_kernel(const float* __restrict__ Wq, const float* __restrict__ Wk,
                                const float* __restrict__ Wv, float* __restrict__ lin){
  int i = blockIdx.x*256 + threadIdx.x;
  if(i >= CC*NQKV) return;
  int c = i / NQKV, n = i % NQKV;
  int h = n / 192, r = n % 192, which = r / 64, d = r % 64;
  const float* W = (which==0)?Wq:((which==1)?Wk:Wv);
  lin[i] = W[((size_t)h*CC + c)*HSZ + d];
}

__global__ void pack_pad_kernel(const float* __restrict__ W, float* __restrict__ lin,
                                int Ks,int Ns,int Kd,int Nd){
  int i = blockIdx.x*256 + threadIdx.x;
  if(i >= Kd*Nd) return;
  int k = i / Nd, n = i % Nd;
  lin[i] = (k<Ks && n<Ns) ? W[(size_t)k*Ns + n] : 0.f;
}

__global__ void pad_vec_kernel(const float* __restrict__ b, float* __restrict__ out,int n,int nd){
  int i = blockIdx.x*256 + threadIdx.x;
  if(i<nd) out[i] = (i<n) ? b[i] : 0.f;
}

// B-fragment swizzle: sw[((kt*NT+nt)*32+lane)*16+e] = W[k][n] with ISA 16-bit pair layout
__global__ void swizzleB_kernel(const float* __restrict__ W, h16* __restrict__ sw,int K,int N){
  int i = blockIdx.x*256 + threadIdx.x;
  if(i >= K*N) return;
  int e = i & 15, lane = (i>>4)&31, tile = i>>9;
  int NT = N>>4;
  int nt = tile % NT, kt = tile / NT;
  int sub = lane>>4, g = e>>1, odd = e&1;
  int kq = (g<4) ? (sub*8 + 2*g) : (16 + sub*8 + 2*(g-4));
  int k = kt*32 + kq + odd;
  int n = nt*16 + (lane&15);
  sw[i] = (h16)W[(size_t)k*N + n];
}

// ---------------------------------------------------------------------------
// Generic GEMM:  C[M,N] = A[M,K](f16) * Bsw(K,N swizzled f16)  [+bias][+res][relu]
// BM=128, BN=64, 256 threads (8 waves, 4x2), wave tile 32x32, K-step 32.
// A tiles DMA'd into LDS by the Tensor Data Mover (wave 0 issues; TENSORcnt).
// ---------------------------------------------------------------------------
template<bool BIAS,bool RES,bool RELU,bool OUTF,bool OUTH>
__global__ __launch_bounds__(256)
void gemm_kernel(const h16* __restrict__ A, const h16* __restrict__ Bsw,
                 const float* __restrict__ bias, const float* __restrict__ res,
                 float* __restrict__ Cf, h16* __restrict__ Ch,
                 int M, int N, int K){
  __shared__ h16 As[2][128][40];        // pitch 80B -> 16B aligned, conflict-free
  const int tid=threadIdx.x, lane=tid&31, wid=tid>>5;
  const int wm=wid>>1, wn=wid&1;
  const int mBase=blockIdx.y*128, nBase=blockIdx.x*64;
  const int KT=K>>5, NT16=N>>4;
  const int lo=lane&15, sub=lane>>4, kb=sub*8;

  v8f acc[2][2]; acc[0][0]=zero8(); acc[0][1]=zero8(); acc[1][0]=zero8(); acc[1][1]=zero8();

  const unsigned ldsB0 = (unsigned)(uintptr_t)&As[0][0][0];
  const unsigned ldsB1 = (unsigned)(uintptr_t)&As[1][0][0];

  if(wid==0){
    tdm_load_tileA(A + (size_t)mBase*K, ldsB0, K, M);
    __builtin_amdgcn_s_wait_tensorcnt(0);
  }
  __syncthreads();

  for(int kt=0; kt<KT; ++kt){
    const int buf = kt&1;
    const bool more = (kt+1)<KT;
    if(more && wid==0)
      tdm_load_tileA(A + (size_t)mBase*K + (kt+1)*32, buf ? ldsB0 : ldsB1, K, M);

    v16h bf[2], af[2];
    #pragma unroll
    for(int ni=0; ni<2; ++ni){
      int nt = (nBase>>4) + wn*2 + ni;
      const h16* bp = Bsw + ((size_t)(kt*NT16 + nt)*32 + lane)*16;
      bf[ni] = cat8(*(const v8h*)bp, *(const v8h*)(bp+8));
    }
    #pragma unroll
    for(int mi=0; mi<2; ++mi){
      int row = wm*32 + mi*16 + lo;
      af[mi] = cat8(*(const v8h*)&As[buf][row][kb], *(const v8h*)&As[buf][row][16+kb]);
    }
    #pragma unroll
    for(int mi=0; mi<2; ++mi)
      #pragma unroll
      for(int ni=0; ni<2; ++ni)
        acc[mi][ni] = __builtin_amdgcn_wmma_f32_16x16x32_f16(
            false, af[mi], false, bf[ni], (short)0, acc[mi][ni], false, false);
    if(more){
      if(wid==0) __builtin_amdgcn_s_wait_tensorcnt(0);
      __syncthreads();
    }
  }
  // epilogue
  #pragma unroll
  for(int mi=0; mi<2; ++mi){
    #pragma unroll
    for(int ni=0; ni<2; ++ni){
      int n = nBase + wn*32 + ni*16 + lo;
      float bv = BIAS ? bias[n] : 0.f;
      #pragma unroll
      for(int r=0; r<8; ++r){
        int m = mBase + wm*32 + mi*16 + r + 8*sub;
        float v = acc[mi][ni][r] + bv;
        if(RES)  v += res[(size_t)m*N + n];
        if(RELU) v = fmaxf(v, 0.f);
        if(OUTF) Cf[(size_t)m*N + n] = v;
        if(OUTH) Ch[(size_t)m*N + n] = (h16)v;
      }
    }
  }
}

// ---------------------------------------------------------------------------
// Flash attention: one block per (qblock64, head, batch). 128 threads, 4 waves,
// each wave owns 16 query rows. K/V^T staged in LDS; P restaged via LDS.
// ---------------------------------------------------------------------------
__global__ __launch_bounds__(128)
void attn_kernel(const h16* __restrict__ qkv, h16* __restrict__ out){
  const int qb=blockIdx.x, hh=blockIdx.y, b=blockIdx.z;
  const int tid=threadIdx.x, lane=tid&31, wid=tid>>5;
  const int lo=lane&15, sub=lane>>4, kb=sub*8;
  __shared__ h16 Qs[64][72];
  __shared__ h16 Ks[64][72];
  __shared__ h16 Vt[64][72];   // Vt[d][s]
  __shared__ h16 Ps[4][16][72];
  const size_t base = ((size_t)b*TT)*NQKV + hh*192;

  #pragma unroll
  for(int j=0;j<4;++j){
    int c=tid+128*j; int r=c>>3, ch=c&7;
    v8h v = *(const v8h*)(qkv + base + (size_t)(qb*64+r)*NQKV + ch*8);
    *(v8h*)&Qs[r][ch*8] = v;
  }
  __syncthreads();
  v16h qa[2];
  #pragma unroll
  for(int w=0;w<2;++w){
    int r = wid*16 + lo;
    qa[w] = cat8(*(const v8h*)&Qs[r][w*32+kb], *(const v8h*)&Qs[r][w*32+16+kb]);
  }

  v8f o[4]; o[0]=zero8(); o[1]=zero8(); o[2]=zero8(); o[3]=zero8();
  float mrow[8], lrow[8];
  #pragma unroll
  for(int r=0;r<8;++r){ mrow[r]=-1e30f; lrow[r]=0.f; }

  for(int sb=0; sb<=qb; ++sb){
    __syncthreads();
    #pragma unroll
    for(int j=0;j<4;++j){
      int c=tid+128*j; int r=c>>3, ch=c&7;
      v8h kv = *(const v8h*)(qkv + base + (size_t)(sb*64+r)*NQKV + 64 + ch*8);
      *(v8h*)&Ks[r][ch*8] = kv;
      v8h vv = *(const v8h*)(qkv + base + (size_t)(sb*64+r)*NQKV + 128 + ch*8);
      #pragma unroll
      for(int e=0;e<8;++e) Vt[ch*8+e][r] = vv[e];
    }
    __syncthreads();

    v8f s[4]; s[0]=zero8(); s[1]=zero8(); s[2]=zero8(); s[3]=zero8();
    #pragma unroll
    for(int ni=0; ni<4; ++ni){
      #pragma unroll
      for(int w=0; w<2; ++w){
        int sc = ni*16 + lo;
        v16h bf = cat8(*(const v8h*)&Ks[sc][w*32+kb], *(const v8h*)&Ks[sc][w*32+16+kb]);
        s[ni] = __builtin_amdgcn_wmma_f32_16x16x32_f16(false, qa[w], false, bf,
                                                       (short)0, s[ni], false, false);
      }
    }
    const bool diag = (sb==qb);
    #pragma unroll
    for(int ni=0; ni<4; ++ni)
      #pragma unroll
      for(int r=0; r<8; ++r){
        float v = s[ni][r]*0.125f;
        if(diag){
          int t = qb*64 + wid*16 + r + 8*sub;
          int ss = sb*64 + ni*16 + lo;
          if(ss > t) v = -1e30f;
        }
        s[ni][r] = v;
      }
    #pragma unroll
    for(int r=0; r<8; ++r){
      float mx = fmaxf(fmaxf(s[0][r],s[1][r]), fmaxf(s[2][r],s[3][r]));
      #pragma unroll
      for(int d=1; d<16; d<<=1) mx = fmaxf(mx, __shfl_xor(mx, d, 16));
      float mnew = fmaxf(mrow[r], mx);
      float al = __expf(mrow[r]-mnew);
      float rs = 0.f;
      #pragma unroll
      for(int ni=0; ni<4; ++ni){ float p=__expf(s[ni][r]-mnew); s[ni][r]=p; rs+=p; }
      #pragma unroll
      for(int d=1; d<16; d<<=1) rs += __shfl_xor(rs, d, 16);
      lrow[r] = lrow[r]*al + rs;  mrow[r] = mnew;
      o[0][r]*=al; o[1][r]*=al; o[2][r]*=al; o[3][r]*=al;
    }
    #pragma unroll
    for(int ni=0; ni<4; ++ni)
      #pragma unroll
      for(int r=0; r<8; ++r)
        Ps[wid][r+8*sub][ni*16+lo] = (h16)s[ni][r];
    __syncthreads();
    #pragma unroll
    for(int w=0; w<2; ++w){
      v16h pa = cat8(*(const v8h*)&Ps[wid][lo][w*32+kb], *(const v8h*)&Ps[wid][lo][w*32+16+kb]);
      #pragma unroll
      for(int di=0; di<4; ++di){
        int dc = di*16 + lo;
        v16h vb = cat8(*(const v8h*)&Vt[dc][w*32+kb], *(const v8h*)&Vt[dc][w*32+16+kb]);
        o[di] = __builtin_amdgcn_wmma_f32_16x16x32_f16(false, pa, false, vb,
                                                       (short)0, o[di], false, false);
      }
    }
  }
  #pragma unroll
  for(int di=0; di<4; ++di)
    #pragma unroll
    for(int r=0; r<8; ++r){
      int t = qb*64 + wid*16 + r + 8*sub;
      int d = di*16 + lo;
      out[((size_t)(b*TT+t))*CC + hh*64 + d] = (h16)(o[di][r]/lrow[r]);
    }
}

// ---------------------------------------------------------------------------
// Newton–Schulz polar (orth) on 384x384 fp32 with v_wmma_f32_16x16x4_f32.
// ---------------------------------------------------------------------------
__global__ __launch_bounds__(256)
void frob_kernel(const float* __restrict__ W, int n, float* __restrict__ invn){
  float s=0.f;
  for(int i=threadIdx.x; i<n; i+=256){ float v=W[i]; s+=v*v; }
  #pragma unroll
  for(int d=1; d<32; d<<=1) s += __shfl_xor(s, d, 32);
  __shared__ float sb[8];
  if((threadIdx.x&31)==0) sb[threadIdx.x>>5]=s;
  __syncthreads();
  if(threadIdx.x==0){
    float t=0.f;
    #pragma unroll
    for(int i=0;i<8;++i) t+=sb[i];
    invn[0] = 0.99f * rsqrtf(t);
  }
}

__global__ void scale_kernel(const float* __restrict__ W, const float* __restrict__ invn,
                             float* __restrict__ X, int n){
  int i = blockIdx.x*256 + threadIdx.x;
  if(i<n) X[i] = W[i]*invn[0];
}

// C = A * B(^T if TRANSB); if AXPBY: Out = 1.5*Xold - 0.5*C  else Out = C
template<bool TRANSB,bool AXPBY>
__global__ __launch_bounds__(256)
void ns_gemm_kernel(const float* __restrict__ A, const float* __restrict__ Bm,
                    const float* __restrict__ Xold, float* __restrict__ Out){
  const int lane=threadIdx.x&31, wid=threadIdx.x>>5;
  const int tile=blockIdx.x*8+wid;
  const int mt=tile/24, nt=tile%24;
  const int lo=lane&15, sub=lane>>4;
  const int mrow=mt*16+lo, ncol=nt*16+lo;
  v8f acc = zero8();
  for(int k=0;k<CC;k+=4){
    int kk = k + 2*sub;
    v2f a = *(const v2f*)(A + (size_t)mrow*CC + kk);
    v2f b;
    if(TRANSB){
      b = *(const v2f*)(Bm + (size_t)ncol*CC + kk);
    }else{
      b[0] = Bm[(size_t)kk*CC + ncol];
      b[1] = Bm[(size_t)(kk+1)*CC + ncol];
    }
    acc = __builtin_amdgcn_wmma_f32_16x16x4_f32(false, a, false, b, (short)0, acc, false, false);
  }
  #pragma unroll
  for(int r=0;r<8;++r){
    int m = mt*16 + r + 8*sub, n = nt*16 + lo;
    float v = acc[r];
    if(AXPBY) v = 1.5f*Xold[(size_t)m*CC+n] - 0.5f*v;
    Out[(size_t)m*CC+n] = v;
  }
}

// ---------------------------------------------------------------------------
extern "C" void kernel_launch(void* const* d_in, const int* in_sizes, int n_in,
                              void* d_out, int out_size, void* d_ws, size_t ws_size,
                              hipStream_t stream){
  (void)in_sizes; (void)n_in; (void)out_size; (void)ws_size;
  const float* x     =(const float*)d_in[0];
  const float* Wq    =(const float*)d_in[1];
  const float* Wk    =(const float*)d_in[2];
  const float* Wv    =(const float*)d_in[3];
  const float* Wp    =(const float*)d_in[4];
  const float* bp    =(const float*)d_in[5];
  const float* scale1=(const float*)d_in[6];
  const float* scale2=(const float*)d_in[7];
  const float* W1    =(const float*)d_in[8];
  const float* b1    =(const float*)d_in[9];
  const float* W2    =(const float*)d_in[10];
  const float* b2    =(const float*)d_in[11];
  const float* Woin  =(const float*)d_in[12];
  const float* Woout =(const float*)d_in[13];

  char* ws=(char*)d_ws; size_t off=0;
  auto alloc=[&](size_t n)->char*{ char* p=ws+off; off=(off+n+255)&~(size_t)255; return p; };

  float* h1f = (float*)alloc((size_t)BTN*CC*4);
  float* h2f = (float*)alloc((size_t)BTN*CC*4);
  h16*  h1h  = (h16*) alloc((size_t)BTN*CC*2);
  char* qr   =        alloc((size_t)BTN*NQKV*2);
  h16*  qkvh = (h16*)qr;
  // weight scratch
  float* wqkv_lin=(float*)alloc((size_t)CC*NQKV*4);
  h16*  wqkv_sw =(h16*) alloc((size_t)CC*NQKV*2);
  h16*  wp_sw   =(h16*) alloc((size_t)CC*CC*2);
  float* w1_lin =(float*)alloc((size_t)CC*PPAD*4);
  h16*  w1_sw   =(h16*) alloc((size_t)CC*PPAD*2);
  float* w2_lin =(float*)alloc((size_t)PPAD*CC*4);
  h16*  w2_sw   =(h16*) alloc((size_t)PPAD*CC*2);
  float* b1p    =(float*)alloc(PPAD*4);
  h16*  oin_sw  =(h16*) alloc((size_t)CC*CC*2);
  h16*  oout_sw =(h16*) alloc((size_t)CC*CC*2);
  float* nsX1   =(float*)alloc((size_t)CC*CC*4);
  float* nsX2   =(float*)alloc((size_t)CC*CC*4);
  float* nsT1   =(float*)alloc((size_t)CC*CC*4);
  float* invn   =(float*)alloc(256);
  // buffer aliases (lifetimes are disjoint)
  h16*  attnh = h1h;
  h16*  h3h   = (h16*)qr;
  h16*  h4h   = (h16*)(qr + (size_t)BTN*CC*2);
  h16*  f1h   = (h16*)(qr + (size_t)BTN*CC*4);
  h16*  f2h   = (h16*)h2f;
  float* f2f  = h1f;

  const int NM = CC*CC; // 147456

  // --- weight prep ---
  pack_qkv_kernel<<<(CC*NQKV+255)/256,256,0,stream>>>(Wq,Wk,Wv,wqkv_lin);
  pack_pad_kernel<<<(CC*PPAD+255)/256,256,0,stream>>>(W1,w1_lin,CC,PP,CC,PPAD);
  pack_pad_kernel<<<(PPAD*CC+255)/256,256,0,stream>>>(W2,w2_lin,PP,CC,PPAD,CC);
  pad_vec_kernel<<<2,256,0,stream>>>(b1,b1p,PP,PPAD);
  swizzleB_kernel<<<(CC*NQKV+255)/256,256,0,stream>>>(wqkv_lin,wqkv_sw,CC,NQKV);
  swizzleB_kernel<<<(NM+255)/256,256,0,stream>>>(Wp,wp_sw,CC,CC);
  swizzleB_kernel<<<(CC*PPAD+255)/256,256,0,stream>>>(w1_lin,w1_sw,CC,PPAD);
  swizzleB_kernel<<<(PPAD*CC+255)/256,256,0,stream>>>(w2_lin,w2_sw,PPAD,CC);

  // --- orth via Newton–Schulz polar iteration (f32 WMMA) ---
  const float* srcs[2] = { Woin, Woout };
  h16* dsts[2] = { oin_sw, oout_sw };
  for(int m=0;m<2;++m){
    frob_kernel<<<1,256,0,stream>>>(srcs[m], NM, invn);
    scale_kernel<<<(NM+255)/256,256,0,stream>>>(srcs[m], invn, nsX1, NM);
    float* Xc=nsX1; float* Xn=nsX2;
    for(int it=0; it<30; ++it){
      ns_gemm_kernel<true ,false><<<72,256,0,stream>>>(Xc, Xc, nullptr, nsT1); // T1 = X X^T
      ns_gemm_kernel<false,true ><<<72,256,0,stream>>>(nsT1, Xc, Xc, Xn);      // X' = 1.5X-0.5 T1 X
      float* t=Xc; Xc=Xn; Xn=t;
    }
    swizzleB_kernel<<<(NM+255)/256,256,0,stream>>>(Xc, dsts[m], CC, CC);
  }

  // --- forward pass ---
  rmsnorm_kernel<<<BTN,128,0,stream>>>(x, scale1, h1f, h1h);                     // ln1
  {
    dim3 g(NQKV/64, BTN/128);
    gemm_kernel<false,false,false,false,true><<<g,256,0,stream>>>(
        h1h, wqkv_sw, nullptr, nullptr, nullptr, qkvh, BTN, NQKV, CC);           // QKV
  }
  attn_kernel<<<dim3(TT/64,HH,BB),128,0,stream>>>(qkvh, attnh);                  // MHA
  {
    dim3 g(CC/64, BTN/128);
    gemm_kernel<true,true,false,true,false><<<g,256,0,stream>>>(
        attnh, wp_sw, bp, h1f, h2f, nullptr, BTN, CC, CC);                       // proj+bp+res
  }
  rmsnorm_kernel<<<BTN,128,0,stream>>>(h2f, scale2, nullptr, h3h);               // ln2
  {
    dim3 g(CC/64, BTN/128);
    gemm_kernel<false,false,false,false,true><<<g,256,0,stream>>>(
        h3h, oin_sw, nullptr, nullptr, nullptr, h4h, BTN, CC, CC);               // @orth(Wo_in)
  }
  {
    dim3 g(PPAD/64, BTN/128);
    gemm_kernel<true,false,true,false,true><<<g,256,0,stream>>>(
        h4h, w1_sw, b1p, nullptr, nullptr, f1h, BTN, PPAD, CC);                  // relu(@W1+b1)
  }
  {
    dim3 g(CC/64, BTN/128);
    gemm_kernel<true,false,false,true,true><<<g,256,0,stream>>>(
        f1h, w2_sw, b2, nullptr, f2f, f2h, BTN, CC, PPAD);                       // @W2+b2
  }
  {
    dim3 g(CC/64, BTN/128);
    gemm_kernel<false,true,false,true,false><<<g,256,0,stream>>>(
        f2h, oout_sw, nullptr, f2f, (float*)d_out, nullptr, BTN, CC, CC);        // @orth(Wo_out)+res
  }
}
